// FastSpeech_36197984371188
// MI455X (gfx1250) — matrix operand order repaired and checked by reading, not compiled, not run on hardware
//
#include <hip/hip_runtime.h>
#include <hip/hip_bf16.h>

// ---------------- model constants ----------------
#define Bsz   16
#define Ssz   256
#define Hsz   384
#define H2sz  1536
#define NHsz  2
#define DHsz  192      // Hsz / NHsz
#define Lsz   4
#define Vsz   256
#define TOUTs 1024
#define MELs  80

typedef __attribute__((ext_vector_type(16))) _Float16 v16h;
typedef __attribute__((ext_vector_type(8)))  float    v8f;
typedef __attribute__((ext_vector_type(4)))  float    f32x4;
typedef __attribute__((ext_vector_type(4)))  unsigned int u32x4;

union Frag16 {                 // 16 f16 = one WMMA A/B fragment (wave32)
    v16h     v;
    _Float16 h[16];
    u32x4    q[2];
};

static __device__ __forceinline__ v8f wmma_f16(v16h a, v16h b, v8f c) {
    // D = A(16x32 f16) * B(32x16 f16) + C(16x16 f32)
    return __builtin_amdgcn_wmma_f32_16x16x32_f16(false, a, false, b,
                                                  (short)0, c, false, false);
}

// =====================================================================
// WMMA GEMM (f16 in, f32 accum):
//   out[M,N] = act( A[M,Kdim] * W[Kdim,N] + bias ),  optional f16 shadow out
// CONV==1: implicit im2col, A is [.,T,Cin], Kdim = 3*Cin (SAME, K=3)
// Block 256 thr (8 waves); tile 128(M) x 128(N) x 32(K); wave tile 32x64.
// A tile staged with global_load_async_to_lds_b128 (ASYNCcnt).
// =====================================================================
template <int CONV>
__global__ __launch_bounds__(256)
void gemm_wmma_kernel(const _Float16* __restrict__ A, const _Float16* __restrict__ W,
                      const float* __restrict__ bias,
                      float* __restrict__ out, _Float16* __restrict__ out16,
                      int M, int N, int Kdim, int T, int Cin, int relu)
{
    __shared__ _Float16 lA[128][32];   // [m][k]  (row-major, async-copy target)
    __shared__ _Float16 lB[128][32];   // [n][k]  (W transposed at stage time)

    const int tid = threadIdx.x;
    const int m0 = blockIdx.y * 128;
    const int n0 = blockIdx.x * 128;

    v8f acc[2][4] = {};

    const int wave = tid >> 5, lane = tid & 31;
    const int wm = (wave >> 1) << 5;      // 0,32,64,96
    const int wn = (wave & 1) << 6;       // 0,64
    const int lm = lane & 15, lg = lane >> 4;

    for (int k0 = 0; k0 < Kdim; k0 += 32) {
        // ---- stage A tile: 128 rows x 32 f16 = 512 x 16B segments, 2/thread,
        //      pure byte copy -> async global->LDS; boundary lanes zero-fill ----
#pragma unroll
        for (int it = 0; it < 2; ++it) {
            const int task = tid + it * 256;
            const int r    = task >> 2;        // 0..127
            const int seg  = task & 3;         // 8 f16 each
            const int m    = m0 + r;
            const int kk   = k0 + seg * 8;
            const unsigned lds = (unsigned)(size_t)&lA[r][seg * 8];
            bool valid = (m < M);
            const _Float16* gp;
            if (CONV) {
                const int tap = kk / Cin;
                const int cin = kk - tap * Cin;
                const int tt  = (m % T) + tap - 1;
                valid = valid && (tt >= 0) && (tt < T);
                gp = A + (size_t)(m + tap - 1) * Cin + cin;
            } else {
                gp = A + (size_t)m * Kdim + kk;
            }
            if (valid) {
                asm volatile("global_load_async_to_lds_b128 %0, %1, off"
                             :: "v"(lds), "v"(gp) : "memory");
            } else {
                const u32x4 z = {0u, 0u, 0u, 0u};
                *(u32x4*)&lA[r][seg * 8] = z;
            }
        }
        // ---- stage B tile transposed: lB[n][k] = W[k][n] (f16 weights) ----
        {
            const int kr = tid >> 3;          // 0..31
            const int ng = (tid & 7) << 4;    // 0..112
            const _Float16* wrow = W + (size_t)(k0 + kr) * N + n0 + ng;
            _Float16 t16[16];
            if (n0 + ng + 16 <= N) {
                ((u32x4*)t16)[0] = *(const u32x4*)wrow;
                ((u32x4*)t16)[1] = *(const u32x4*)(wrow + 8);
            } else {
#pragma unroll
                for (int j = 0; j < 16; ++j) {
                    const int n = n0 + ng + j;
                    t16[j] = (n < N) ? wrow[j] : (_Float16)0.f;
                }
            }
#pragma unroll
            for (int j = 0; j < 16; ++j) lB[ng + j][kr] = t16[j];
        }
        asm volatile("s_wait_asynccnt 0" ::: "memory");
        __syncthreads();

        // ---- wave compute: 2(M) x 4(N) subtiles of 16x16, 8 WMMA/chunk ----
        Frag16 af[2], bf[4];
#pragma unroll
        for (int s = 0; s < 2; ++s) {
            const int mr = wm + s * 16 + lm;
            af[s].q[0] = *(const u32x4*)&lA[mr][lg * 16];
            af[s].q[1] = *(const u32x4*)&lA[mr][lg * 16 + 8];
        }
#pragma unroll
        for (int s = 0; s < 4; ++s) {
            const int nr = wn + s * 16 + lm;
            bf[s].q[0] = *(const u32x4*)&lB[nr][lg * 16];
            bf[s].q[1] = *(const u32x4*)&lB[nr][lg * 16 + 8];
        }
#pragma unroll
        for (int sm = 0; sm < 2; ++sm)
#pragma unroll
            for (int sn = 0; sn < 4; ++sn)
                acc[sm][sn] = wmma_f16(af[sm].v, bf[sn].v, acc[sm][sn]);
        __syncthreads();
    }

    // ---- epilogue: bias (+relu); f32 and/or f16 stores in C layout ----
#pragma unroll
    for (int sm = 0; sm < 2; ++sm) {
#pragma unroll
        for (int sn = 0; sn < 4; ++sn) {
            const int n = n0 + wn + sn * 16 + lm;
            const int mbase = m0 + wm + sm * 16;
            if (n < N) {
                const float bv = bias ? bias[n] : 0.f;
#pragma unroll
                for (int v = 0; v < 8; ++v) {
                    const int m = mbase + v + 8 * lg;
                    if (m < M) {
                        float x = acc[sm][sn][v] + bv;
                        if (relu) x = fmaxf(x, 0.f);
                        if (out)   out[(size_t)m * N + n] = x;
                        if (out16) out16[(size_t)m * N + n] = (_Float16)x;
                    }
                }
            }
        }
    }
}

// =====================================================================
// Flash attention (wave32 WMMA), all-f16 operands, f32 accumulate.
// One wave per (b, h, 16-query tile). Q,K,V,O: [B,T,H] f16, head at h*DH.
// =====================================================================
__global__ __launch_bounds__(32)
void attn_flash_kernel(const _Float16* __restrict__ Q, const _Float16* __restrict__ K,
                       const _Float16* __restrict__ V, _Float16* __restrict__ O, int T)
{
    __shared__ _Float16 ldsVt[DHsz][32];  // [d][key]  12 KB
    __shared__ _Float16 ldsP[16][32];     // [m][k]     1 KB

    const int b  = blockIdx.z;
    const int h  = blockIdx.y;
    const int q0 = blockIdx.x * 16;
    const int lane = threadIdx.x;
    const int lm = lane & 15, lg = lane >> 4;
    const float rscale = 0.07216878364870323f;  // 1/sqrt(192)

    // ---- resident Q fragments over dh = 6 chunks of 32 (direct f16 loads) ----
    Frag16 qf[6];
    {
        const _Float16* qbase = Q + (size_t)(b * T + q0 + lm) * Hsz + h * DHsz + lg * 16;
#pragma unroll
        for (int c = 0; c < 6; ++c) {
            qf[c].q[0] = *(const u32x4*)(qbase + c * 32);
            qf[c].q[1] = *(const u32x4*)(qbase + c * 32 + 8);
        }
    }

    v8f oacc[12] = {};
    float rmax[8], rsum[8];
#pragma unroll
    for (int v = 0; v < 8; ++v) { rmax[v] = -1e30f; rsum[v] = 0.f; }

    for (int j0 = 0; j0 < T; j0 += 32) {
        // ---- stage V chunk transposed in LDS: ldsVt[d][key] ----
        {
            const _Float16* vrow = V + (size_t)(b * T + j0 + lane) * Hsz + h * DHsz;
#pragma unroll
            for (int d0 = 0; d0 < DHsz; d0 += 8) {
                _Float16 t8[8];
                *(u32x4*)t8 = *(const u32x4*)(vrow + d0);
#pragma unroll
                for (int e = 0; e < 8; ++e) ldsVt[d0 + e][lane] = t8[e];
            }
        }
        __syncthreads();

        // ---- S = Q * K^T for 32 keys (two 16x16 tiles) ----
        v8f s0 = {}, s1 = {};
#pragma unroll
        for (int c = 0; c < 6; ++c) {
            Frag16 kf0, kf1;
            const _Float16* p0 = K + (size_t)(b * T + j0 + lm) * Hsz + h * DHsz
                                 + c * 32 + lg * 16;
            const _Float16* p1 = p0 + (size_t)16 * Hsz;
            kf0.q[0] = *(const u32x4*)p0;  kf0.q[1] = *(const u32x4*)(p0 + 8);
            kf1.q[0] = *(const u32x4*)p1;  kf1.q[1] = *(const u32x4*)(p1 + 8);
            s0 = wmma_f16(qf[c].v, kf0.v, s0);
            s1 = wmma_f16(qf[c].v, kf1.v, s1);
        }

        // ---- online softmax (rows per-VGPR; 16-lane half reductions) ----
        float corr[8];
#pragma unroll
        for (int v = 0; v < 8; ++v) {
            float a  = s0[v] * rscale;
            float c2 = s1[v] * rscale;
            float mx = fmaxf(a, c2);
#pragma unroll
            for (int d = 1; d < 16; d <<= 1) mx = fmaxf(mx, __shfl_xor(mx, d, 32));
            const float nm = fmaxf(rmax[v], mx);
            corr[v] = __expf(rmax[v] - nm);
            rmax[v] = nm;
            const float p0 = __expf(a - nm);
            const float p1 = __expf(c2 - nm);
            float ps = p0 + p1;
#pragma unroll
            for (int d = 1; d < 16; d <<= 1) ps += __shfl_xor(ps, d, 32);
            rsum[v] = rsum[v] * corr[v] + ps;
            ldsP[v + 8 * lg][lm]      = (_Float16)p0;
            ldsP[v + 8 * lg][16 + lm] = (_Float16)p1;
        }
#pragma unroll
        for (int dt = 0; dt < 12; ++dt)
#pragma unroll
            for (int v = 0; v < 8; ++v) oacc[dt][v] *= corr[v];
        __syncthreads();

        // ---- O += P(16x32) * V(32x16) per d-tile (12 WMMA) ----
        Frag16 pf;
        pf.q[0] = *(const u32x4*)&ldsP[lm][lg * 16];
        pf.q[1] = *(const u32x4*)&ldsP[lm][lg * 16 + 8];
#pragma unroll
        for (int dt = 0; dt < 12; ++dt) {
            Frag16 vf;
            vf.q[0] = *(const u32x4*)&ldsVt[dt * 16 + lm][lg * 16];
            vf.q[1] = *(const u32x4*)&ldsVt[dt * 16 + lm][lg * 16 + 8];
            oacc[dt] = wmma_f16(pf.v, vf.v, oacc[dt]);
        }
        __syncthreads();
    }

    // ---- normalize + store f16 (consumed by O-projection GEMM) ----
#pragma unroll
    for (int dt = 0; dt < 12; ++dt) {
        const int d = dt * 16 + lm;
#pragma unroll
        for (int v = 0; v < 8; ++v) {
            const int m = q0 + v + 8 * lg;
            O[(size_t)(b * T + m) * Hsz + h * DHsz + d] =
                (_Float16)(oacc[dt][v] / rsum[v]);
        }
    }
}

// =====================================================================
// Small fused kernels
// =====================================================================
__global__ void cvt_f16_kernel(const float* __restrict__ src,
                               _Float16* __restrict__ dst, int n)
{
    const int i = blockIdx.x * 256 + threadIdx.x;
    if (i < n) dst[i] = (_Float16)src[i];
}

__global__ void embed_kernel(const int* __restrict__ tokens,
                             const float* __restrict__ emb, float* __restrict__ x)
{
    const int row = blockIdx.x;
    const int tok = tokens[row];
    for (int c = threadIdx.x; c < Hsz; c += 128)
        x[(size_t)row * Hsz + c] = emb[(size_t)tok * Hsz + c] * 19.595917942265423f;
}

// x += positional encoding; also writes f16 shadow
__global__ void posenc_kernel(float* __restrict__ x, _Float16* __restrict__ x16, int T)
{
    const int row = blockIdx.x;
    const int t = row % T;
    const int i = threadIdx.x;  // 192
    const float den = __expf(-9.210340371976184f * (2.f * i) / (float)Hsz);
    const float ang = (float)t * den;
    const float v0 = x[(size_t)row * Hsz + 2 * i]     + __sinf(ang);
    const float v1 = x[(size_t)row * Hsz + 2 * i + 1] + __cosf(ang);
    x[(size_t)row * Hsz + 2 * i]     = v0;
    x[(size_t)row * Hsz + 2 * i + 1] = v1;
    x16[(size_t)row * Hsz + 2 * i]     = (_Float16)v0;
    x16[(size_t)row * Hsz + 2 * i + 1] = (_Float16)v1;
}

// out = LayerNorm(x (+ r)) * sb[0:H] + sb[H:2H]; f32 out + f16 shadow
__global__ void add_ln_kernel(const float* __restrict__ x, const float* __restrict__ r,
                              const float* __restrict__ sb,
                              float* __restrict__ out, _Float16* __restrict__ out16)
{
    const int row = blockIdx.x;
    const int tid = threadIdx.x;
    float vals[3];
    float s = 0.f, ss = 0.f;
#pragma unroll
    for (int i = 0; i < 3; ++i) {
        const int c = tid + i * 128;
        float v = x[(size_t)row * Hsz + c];
        if (r) v += r[(size_t)row * Hsz + c];
        vals[i] = v; s += v; ss += v * v;
    }
    __shared__ float red0[128], red1[128];
    red0[tid] = s; red1[tid] = ss;
    __syncthreads();
    for (int off = 64; off > 0; off >>= 1) {
        if (tid < off) { red0[tid] += red0[tid + off]; red1[tid] += red1[tid + off]; }
        __syncthreads();
    }
    const float mean = red0[0] * (1.f / Hsz);
    const float var  = red1[0] * (1.f / Hsz) - mean * mean;
    const float inv  = rsqrtf(var + 1e-5f);
#pragma unroll
    for (int i = 0; i < 3; ++i) {
        const int c = tid + i * 128;
        const float o = (vals[i] - mean) * inv * sb[c] + sb[Hsz + c];
        out[(size_t)row * Hsz + c] = o;
        out16[(size_t)row * Hsz + c] = (_Float16)o;
    }
}

__global__ void cumsum_kernel(const int* __restrict__ dur, int* __restrict__ cum)
{
    const int b = blockIdx.x;
    if (threadIdx.x == 0) {
        int run = 0;
        for (int s = 0; s < Ssz; ++s) { run += dur[b * Ssz + s]; cum[b * Ssz + s] = run; }
    }
}

__global__ void regulate_kernel(const float* __restrict__ x, const int* __restrict__ cum,
                                float* __restrict__ y)
{
    const int b = blockIdx.y, t = blockIdx.x, tid = threadIdx.x;
    __shared__ int sidx, sval;
    if (tid == 0) {
        const int* c = cum + b * Ssz;
        int lo = 0, hi = Ssz;
        while (lo < hi) { const int mid = (lo + hi) >> 1; if (c[mid] <= t) lo = mid + 1; else hi = mid; }
        sval = (t < c[Ssz - 1]) ? 1 : 0;
        sidx = lo < (Ssz - 1) ? lo : (Ssz - 1);
    }
    __syncthreads();
    const int idx = sidx, valid = sval;
    for (int c2 = tid; c2 < Hsz; c2 += 128)
        y[((size_t)b * TOUTs + t) * Hsz + c2] =
            valid ? x[((size_t)b * Ssz + idx) * Hsz + c2] : 0.f;
}

__global__ void rowdot_kernel(const float* __restrict__ h, const float* __restrict__ w,
                              const float* __restrict__ b, float* __restrict__ out, int Mrows)
{
    const int wave = threadIdx.x >> 5, lane = threadIdx.x & 31;
    const int row = blockIdx.x * 8 + wave;
    if (row >= Mrows) return;
    float s = 0.f;
    for (int c = lane; c < Hsz; c += 32) s += h[(size_t)row * Hsz + c] * w[c];
#pragma unroll
    for (int m = 16; m > 0; m >>= 1) s += __shfl_xor(s, m, 32);
    if (lane == 0) out[row] = s + b[0];
}

__global__ void mel_tr_kernel(const float* __restrict__ melt, float* __restrict__ out)
{
    const int idx = blockIdx.x * 256 + threadIdx.x;
    const int total = Bsz * MELs * TOUTs;
    if (idx >= total) return;
    const int t = idx % TOUTs;
    const int c = (idx / TOUTs) % MELs;
    const int b = idx / (MELs * TOUTs);
    out[idx] = melt[((size_t)b * TOUTs + t) * MELs + c];
}

// =====================================================================
// Orchestration
// =====================================================================
extern "C" void kernel_launch(void* const* d_in, const int* in_sizes, int n_in,
                              void* d_out, int out_size, void* d_ws, size_t ws_size,
                              hipStream_t stream)
{
    (void)in_sizes; (void)n_in; (void)out_size; (void)ws_size;

    const int*   tokens   = (const int*)  d_in[0];
    const int*   alignes  = (const int*)  d_in[1];
    const float* tok_emb  = (const float*)d_in[2];
    const float* attn_w[2] = { (const float*)d_in[3],  (const float*)d_in[10] };
    const float* attn_b[2] = { (const float*)d_in[4],  (const float*)d_in[11] };
    const float* c1_w[2]   = { (const float*)d_in[5],  (const float*)d_in[12] };
    const float* c1_b[2]   = { (const float*)d_in[6],  (const float*)d_in[13] };
    const float* c2_w[2]   = { (const float*)d_in[7],  (const float*)d_in[14] };
    const float* c2_b[2]   = { (const float*)d_in[8],  (const float*)d_in[15] };
    const float* lnp[2]    = { (const float*)d_in[9],  (const float*)d_in[16] };
    const float* al_c1w = (const float*)d_in[17];
    const float* al_c1b = (const float*)d_in[18];
    const float* al_c2w = (const float*)d_in[19];
    const float* al_c2b = (const float*)d_in[20];
    const float* al_ln  = (const float*)d_in[21];
    const float* al_lw  = (const float*)d_in[22];
    const float* al_lb  = (const float*)d_in[23];
    const float* dl_w   = (const float*)d_in[24];
    const float* dl_b   = (const float*)d_in[25];

    float* out = (float*)d_out;     // [B,80,1024] then [B,S]

    const size_t ME = (size_t)Bsz * Ssz;     // 4096 rows (encoder)
    const size_t MD = (size_t)Bsz * TOUTs;   // 16384 rows (decoder)

    // ---- f32 workspace ----
    float* ws = (float*)d_ws;
    size_t off = 0;
    float* XE   = ws + off; off += ME * Hsz;
    float* XD   = ws + off; off += MD * Hsz;
    float* TMPb = ws + off; off += MD * Hsz;
    float* HB   = ws + off; off += ME * Hsz;
    float* MELT = ws + off; off += MD * MELs;
    int*   CUM  = (int*)(ws + off); off += ME;

    // ---- f16 workspace (activations shadows + converted weights) ----
    _Float16* h = (_Float16*)(ws + off);
    size_t ho = 0;
    _Float16* XE16  = h + ho; ho += ME * Hsz;
    _Float16* XD16  = h + ho; ho += MD * Hsz;
    _Float16* Q16   = h + ho; ho += MD * Hsz;
    _Float16* K16   = h + ho; ho += MD * Hsz;
    _Float16* V16   = h + ho; ho += MD * Hsz;
    _Float16* ATT16 = h + ho; ho += MD * Hsz;
    _Float16* FF116 = h + ho; ho += MD * H2sz;
    _Float16* HB16  = h + ho; ho += ME * Hsz;

    const size_t AWsz = (size_t)Lsz * 4 * Hsz * Hsz;
    const size_t C1sz = (size_t)Lsz * 3 * Hsz * H2sz;
    const size_t C2sz = (size_t)Lsz * 3 * H2sz * Hsz;
    const size_t ALsz = (size_t)3 * Hsz * Hsz;
    _Float16* aw16[2]; _Float16* c1w16[2]; _Float16* c2w16[2];
    for (int w = 0; w < 2; ++w) {
        aw16[w]  = h + ho; ho += AWsz;
        c1w16[w] = h + ho; ho += C1sz;
        c2w16[w] = h + ho; ho += C2sz;
    }
    _Float16* alc1w16 = h + ho; ho += ALsz;
    _Float16* alc2w16 = h + ho; ho += ALsz;
    _Float16* dlw16   = h + ho; ho += (size_t)Hsz * MELs;

    auto cvt = [&](const float* src, _Float16* dst, size_t n) {
        cvt_f16_kernel<<<(int)((n + 255) / 256), 256, 0, stream>>>(src, dst, (int)n);
    };
    for (int w = 0; w < 2; ++w) {
        cvt(attn_w[w], aw16[w], AWsz);
        cvt(c1_w[w], c1w16[w], C1sz);
        cvt(c2_w[w], c2w16[w], C2sz);
    }
    cvt(al_c1w, alc1w16, ALsz);
    cvt(al_c2w, alc2w16, ALsz);
    cvt(dl_w, dlw16, (size_t)Hsz * MELs);

    auto gemm = [&](const _Float16* A, const _Float16* W, const float* bias,
                    float* o, _Float16* o16,
                    int Mr, int Nc, int Kd, int conv, int T, int Cin, int relu) {
        dim3 g((Nc + 127) / 128, (Mr + 127) / 128);
        if (conv)
            gemm_wmma_kernel<1><<<g, 256, 0, stream>>>(A, W, bias, o, o16,
                                                       Mr, Nc, Kd, T, Cin, relu);
        else
            gemm_wmma_kernel<0><<<g, 256, 0, stream>>>(A, W, bias, o, o16,
                                                       Mr, Nc, Kd, T, Cin, relu);
    };

    auto fft_stack = [&](float* X, _Float16* X16, int T, int w) {
        const int Mr = Bsz * T;
        for (int i = 0; i < Lsz; ++i) {
            const _Float16* aw = aw16[w] + (size_t)i * 4 * Hsz * Hsz;
            const float*    ab = attn_b[w] + (size_t)i * 4 * Hsz;
            gemm(X16, aw + 0 * Hsz * Hsz, ab + 0 * Hsz, nullptr, Q16, Mr, Hsz, Hsz, 0, T, Hsz, 0);
            gemm(X16, aw + 1 * Hsz * Hsz, ab + 1 * Hsz, nullptr, K16, Mr, Hsz, Hsz, 0, T, Hsz, 0);
            gemm(X16, aw + 2 * Hsz * Hsz, ab + 2 * Hsz, nullptr, V16, Mr, Hsz, Hsz, 0, T, Hsz, 0);
            attn_flash_kernel<<<dim3(T / 16, NHsz, Bsz), 32, 0, stream>>>(Q16, K16, V16, ATT16, T);
            gemm(ATT16, aw + 3 * Hsz * Hsz, ab + 3 * Hsz, TMPb, nullptr, Mr, Hsz, Hsz, 0, T, Hsz, 0);
            add_ln_kernel<<<Mr, 128, 0, stream>>>(X, TMPb,
                lnp[w] + (size_t)(i * 2 + 0) * 2 * Hsz, X, X16);
            gemm(X16, c1w16[w] + (size_t)i * 3 * Hsz * H2sz, c1_b[w] + (size_t)i * H2sz,
                 nullptr, FF116, Mr, H2sz, 3 * Hsz, 1, T, Hsz, 1);
            gemm(FF116, c2w16[w] + (size_t)i * 3 * H2sz * Hsz, c2_b[w] + (size_t)i * Hsz,
                 TMPb, nullptr, Mr, Hsz, 3 * H2sz, 1, T, H2sz, 0);
            add_ln_kernel<<<Mr, 128, 0, stream>>>(X, TMPb,
                lnp[w] + (size_t)(i * 2 + 1) * 2 * Hsz, X, X16);
        }
    };

    // ---- encoder ----
    embed_kernel<<<(int)ME, 128, 0, stream>>>(tokens, tok_emb, XE);
    posenc_kernel<<<(int)ME, 192, 0, stream>>>(XE, XE16, Ssz);
    fft_stack(XE, XE16, Ssz, 0);

    // ---- aligner / duration predictor ----
    gemm(XE16, alc1w16, al_c1b, TMPb, nullptr, (int)ME, Hsz, 3 * Hsz, 1, Ssz, Hsz, 1);
    add_ln_kernel<<<(int)ME, 128, 0, stream>>>(TMPb, nullptr, al_ln + 0 * 2 * Hsz, HB, HB16);
    gemm(HB16, alc2w16, al_c2b, TMPb, nullptr, (int)ME, Hsz, 3 * Hsz, 1, Ssz, Hsz, 1);
    add_ln_kernel<<<(int)ME, 128, 0, stream>>>(TMPb, nullptr, al_ln + 1 * 2 * Hsz, HB, HB16);
    rowdot_kernel<<<((int)ME + 7) / 8, 256, 0, stream>>>(HB, al_lw, al_lb,
        out + (size_t)Bsz * MELs * TOUTs, (int)ME);

    // ---- length regulator ----
    cumsum_kernel<<<Bsz, 32, 0, stream>>>(alignes, CUM);
    regulate_kernel<<<dim3(TOUTs, Bsz), 128, 0, stream>>>(XE, CUM, XD);
    posenc_kernel<<<(int)MD, 192, 0, stream>>>(XD, XD16, TOUTs);

    // ---- decoder ----
    fft_stack(XD, XD16, TOUTs, 1);

    // ---- mel projection + transpose ----
    gemm(XD16, dlw16, dl_b, MELT, nullptr, (int)MD, MELs, Hsz, 0, TOUTs, Hsz, 0);
    const int total = Bsz * MELs * TOUTs;
    mel_tr_kernel<<<(total + 255) / 256, 256, 0, stream>>>(MELT, out);
}
